// PPCNN2_73169062855071
// MI455X (gfx1250) — matrix operand order repaired and checked by reading, not compiled, not run on hardware
//
#include <hip/hip_runtime.h>
#include <hip/hip_bf16.h>
#include <math.h>
#include <stdint.h>

// ============================================================================
// PPCNN (PointNet++-style) forward for MI455X / gfx1250.
//  - All MLP / 1x1-conv layers run through one fused bf16 WMMA GEMM:
//      Y = relu(g*(W@X)+b), f32 accumulate, v_wmma_f32_16x16x32_bf16.
//  - Operands are pre-cast per layer to padded bf16 (W: [Opad,Cpad],
//    X: transposed [Npad,Cpad]) so the GEMM inner loop is branch-free and
//    every WMMA fragment is dword-packed in LDS.
//  - Tile staging uses CDNA5 async copies (global_load_async_to_lds_b128),
//    double-buffered and gated with s_wait_asynccnt.
//  - Irregular ops (FPS, ball query, voxel scatter-mean, 3x3 grid conv, SE,
//    3-NN interpolation, max-over-K) use VALU/LDS/atomic kernels.
// ============================================================================

#define WG 256
static const int Bn = 8;        // batch
static const int NPTS = 8192;   // dense points

typedef __attribute__((ext_vector_type(16))) __bf16    bf16x16;
typedef __attribute__((ext_vector_type(8)))  float     f32x8;
typedef __attribute__((ext_vector_type(4)))  unsigned  u32x4;

__device__ inline unsigned short f2bf(float f) {
  union { float f; unsigned u; } x; x.f = f;
  unsigned r = (x.u + 0x7FFFu + ((x.u >> 16) & 1u)) >> 16;  // RNE
  return (unsigned short)r;
}
__device__ inline long long gtid() {
  return (long long)blockIdx.x * blockDim.x + threadIdx.x;
}

// --- CDNA5 async global->LDS copy (ASYNCcnt-tracked) ------------------------
__device__ inline void async_cp128(void* lds, const void* gptr) {
  unsigned loff = (unsigned)(size_t)lds;  // low 32 bits of generic = LDS offset
  asm volatile("global_load_async_to_lds_b128 %0, %1, off"
               :: "v"(loff), "v"(gptr) : "memory");
}
template <int n> __device__ inline void wait_async() {
  asm volatile("s_wait_asynccnt %0" :: "n"(n) : "memory");
}

// ---------------------------------------------------------------------------
// Per-layer operand casts (padded bf16)
// ---------------------------------------------------------------------------
// Wbf[o, c] (Opad x Cpad), zero padded
__global__ void k_castW(const float* __restrict__ W, unsigned short* __restrict__ Wbf,
                        int O, int C, int Cpad, long long tot) {
  long long g = gtid(); if (g >= tot) return;
  int c = (int)(g % Cpad); int o = (int)(g / Cpad);
  float v = (o < O && c < C) ? W[(size_t)o * C + c] : 0.f;
  Wbf[g] = f2bf(v);
}
// Xt[b, n, c] (Npad x Cpad per batch) = X[b, c, n], zero padded
__global__ void k_castXT(const float* __restrict__ X, unsigned short* __restrict__ Xt,
                         int C, int N, int Cpad, int Npad, long long tot) {
  long long g = gtid(); if (g >= tot) return;
  int c = (int)(g % Cpad); long long t = g / Cpad;
  int n = (int)(t % Npad); int b = (int)(t / Npad);
  float v = (c < C && n < N) ? X[((size_t)b * C + c) * N + n] : 0.f;
  Xt[g] = f2bf(v);
}

// ---------------------------------------------------------------------------
// Fused WMMA GEMM: per batch bz: Y[bz] = act(W @ X[bz])
//   Wbf: [Opad,Cpad] bf16; Xt: [B,Npad,Cpad] bf16 (transposed); Y: [B,O,N] f32
//   mode 1: y = relu(y*G[o]+Bv[o]);  mode 2: y = y + Bv[o]
// 64x64 block tile, K slab 32, 8 waves: wave owns two 16x16 D tiles.
// Double-buffered async tile staging; LDS row pitch 40 ushorts (80 B) for
// conflict-free ds_load_b128 fragment reads.
// ---------------------------------------------------------------------------
__global__ __launch_bounds__(256)
void k_gemm(const unsigned short* __restrict__ Wbf,
            const unsigned short* __restrict__ Xt,
            const float* __restrict__ G, const float* __restrict__ Bv,
            float* __restrict__ Y, int O, int C, int N,
            int Cpad, int Npad, int mode) {
  __shared__ __align__(16) unsigned short As[2][64 * 40];
  __shared__ __align__(16) unsigned short Bs[2][64 * 40];

  const int bz = blockIdx.z;
  const unsigned short* Xb = Xt + (size_t)bz * Npad * Cpad;
  float* Yb = Y + (size_t)bz * O * N;
  const int o0 = blockIdx.y * 64;
  const int n0 = blockIdx.x * 64;
  const int tid  = threadIdx.x;
  const int lane = tid & 31;
  const int wave = tid >> 5;          // 8 waves
  const int wo   = wave & 3;          // O sub-tile
  const int wn0  = (wave >> 2) * 2;   // first of two N sub-tiles
  const int m    = lane & 15;
  const int hi   = lane >> 4;
  const int sr   = tid >> 2;          // staging row 0..63
  const int sq   = tid & 3;           // 16B chunk in row

  f32x8 acc0 = {0.f, 0.f, 0.f, 0.f, 0.f, 0.f, 0.f, 0.f};
  f32x8 acc1 = acc0;

  // prologue: stage slab 0 into buffer 0
  async_cp128(&As[0][sr * 40 + sq * 8], Wbf + (size_t)(o0 + sr) * Cpad + sq * 8);
  async_cp128(&Bs[0][sr * 40 + sq * 8], Xb  + (size_t)(n0 + sr) * Cpad + sq * 8);

  int cur = 0;
  for (int k0 = 0; k0 < Cpad; k0 += 32) {
    const bool more = (k0 + 32) < Cpad;
    if (more) {  // stage next slab into other buffer
      int nk = k0 + 32;
      async_cp128(&As[cur ^ 1][sr * 40 + sq * 8],
                  Wbf + (size_t)(o0 + sr) * Cpad + nk + sq * 8);
      async_cp128(&Bs[cur ^ 1][sr * 40 + sq * 8],
                  Xb + (size_t)(n0 + sr) * Cpad + nk + sq * 8);
      wait_async<2>();  // current slab's 2 async ops complete (in-order)
    } else {
      wait_async<0>();
    }
    __syncthreads();

    union Frag { bf16x16 v; u32x4 q[2]; } A_, B0_, B1_;
    const u32x4* Ar  = (const u32x4*)(&As[cur][(wo * 16 + m) * 40]);
    A_.q[0] = Ar[2 * hi];
    A_.q[1] = Ar[2 * hi + 1];
    const u32x4* Br0 = (const u32x4*)(&Bs[cur][(wn0 * 16 + m) * 40]);
    B0_.q[0] = Br0[2 * hi];
    B0_.q[1] = Br0[2 * hi + 1];
    const u32x4* Br1 = (const u32x4*)(&Bs[cur][((wn0 + 1) * 16 + m) * 40]);
    B1_.q[0] = Br1[2 * hi];
    B1_.q[1] = Br1[2 * hi + 1];

    acc0 = __builtin_amdgcn_wmma_f32_16x16x32_bf16(false, A_.v, false, B0_.v,
                                                   (short)0, acc0, false, false);
    acc1 = __builtin_amdgcn_wmma_f32_16x16x32_bf16(false, A_.v, false, B1_.v,
                                                   (short)0, acc1, false, false);
    __syncthreads();  // done reading `cur` before it is restaged next iter
    cur ^= 1;
  }

  // Epilogue: D layout VGPR v -> M=v+8*hi, N=lane&15
#pragma unroll
  for (int v = 0; v < 8; ++v) {
    int oo = o0 + wo * 16 + v + 8 * hi;
    if (oo >= O) continue;
    float gg = 1.f, bb = 0.f;
    if (mode == 1)      { gg = G[oo]; bb = Bv[oo]; }
    else if (mode == 2) { bb = Bv[oo]; }
    float rA = acc0[v] * gg + bb;
    float rB = acc1[v] * gg + bb;
    if (mode == 1) { rA = fmaxf(rA, 0.f); rB = fmaxf(rB, 0.f); }
    int nnA = n0 + wn0 * 16 + m;
    int nnB = n0 + (wn0 + 1) * 16 + m;
    if (nnA < N) Yb[(size_t)oo * N + nnA] = rA;
    if (nnB < N) Yb[(size_t)oo * N + nnB] = rB;
  }
}

// ---------------------------------------------------------------------------
// Elementwise / gather / scatter helpers
// ---------------------------------------------------------------------------
__global__ void k_zero(float* __restrict__ p, long long n) {
  long long g = gtid(); if (g < n) p[g] = 0.f;
}

// dst[b, cd0+c, n] = src[b, cs0+c, n]
__global__ void k_place(const float* __restrict__ src, float* __restrict__ dst,
                        int Cs, int cs0, int Cd, int cd0, int Cc, int Nn,
                        long long tot) {
  long long g = gtid(); if (g >= tot) return;
  int n = (int)(g % Nn); long long t = g / Nn;
  int c = (int)(t % Cc); int b = (int)(t / Cc);
  dst[((size_t)b * Cd + cd0 + c) * Nn + n] =
      src[((size_t)b * Cs + cs0 + c) * Nn + n];
}

// Farthest point sampling; one block per batch. coords: [B,3,Np]
__global__ void k_fps(const float* __restrict__ coords, int* __restrict__ idx,
                      int Np, int M) {
  __shared__ float dist[NPTS];
  __shared__ float rv[256];
  __shared__ int   ri[256];
  __shared__ int   s_far;
  __shared__ float cx, cy, cz;
  int b = blockIdx.x, t = threadIdx.x;
  const float* px = coords + (size_t)b * 3 * Np;
  const float* py = px + Np;
  const float* pz = py + Np;
  for (int i = t; i < Np; i += 256) dist[i] = 1e10f;
  if (t == 0) s_far = 0;
  __syncthreads();
  for (int mm = 0; mm < M; ++mm) {
    if (t == 0) {
      int far = s_far;
      idx[(size_t)b * M + mm] = far;
      cx = px[far]; cy = py[far]; cz = pz[far];
    }
    __syncthreads();
    float bx = cx, by = cy, bz = cz;
    float bestv = -1.f; int besti = 0;
    for (int i = t; i < Np; i += 256) {
      float dx = px[i] - bx, dy = py[i] - by, dz = pz[i] - bz;
      float dd = fminf(dist[i], dx * dx + dy * dy + dz * dz);
      dist[i] = dd;
      if (dd > bestv) { bestv = dd; besti = i; }
    }
    rv[t] = bestv; ri[t] = besti;
    __syncthreads();
    for (int s = 128; s > 0; s >>= 1) {
      if (t < s && rv[t + s] > rv[t]) { rv[t] = rv[t + s]; ri[t] = ri[t + s]; }
      __syncthreads();
    }
    if (t == 0) s_far = ri[0];
    __syncthreads();
  }
}

// dst[b,c,m] = src[b,c,idx[b,m]]
__global__ void k_gather_cols(const float* __restrict__ src,
                              const int* __restrict__ idx,
                              float* __restrict__ dst,
                              int C, int Np, int M, long long tot) {
  long long g = gtid(); if (g >= tot) return;
  int mm = (int)(g % M); long long t = g / M;
  int c = (int)(t % C); int b = (int)(t / C);
  dst[((size_t)b * C + c) * M + mm] =
      src[((size_t)b * C + c) * Np + idx[(size_t)b * M + mm]];
}

// Ball query: first K indices (ascending n) within radius; fill with first hit.
__global__ void k_ballquery(const float* __restrict__ coords,
                            const float* __restrict__ centers,
                            int* __restrict__ nidx, float r2,
                            int Np, int M, int K, long long tot) {
  long long g = gtid(); if (g >= tot) return;
  int mm = (int)(g % M); int b = (int)(g / M);
  const float* px = coords + (size_t)b * 3 * Np;
  const float* py = px + Np;
  const float* pz = py + Np;
  float cx = centers[((size_t)b * 3 + 0) * M + mm];
  float cy = centers[((size_t)b * 3 + 1) * M + mm];
  float cz = centers[((size_t)b * 3 + 2) * M + mm];
  int* out = nidx + ((size_t)b * M + mm) * K;
  int cnt = 0, first = 0; bool got = false;
  for (int n = 0; n < Np && cnt < K; ++n) {
    float dx = px[n] - cx, dy = py[n] - cy, dz = pz[n] - cz;
    if (dx * dx + dy * dy + dz * dz < r2) {
      if (!got) { got = true; first = n; }
      out[cnt++] = n;
    }
  }
  for (; cnt < K; ++cnt) out[cnt] = first;
}

// out[b, ch, m, k]: ch<3 -> coords diff; else feat gather
__global__ void k_group(const float* __restrict__ feat,
                        const float* __restrict__ coords,
                        const float* __restrict__ centers,
                        const int* __restrict__ nidx,
                        float* __restrict__ out,
                        int C, int Np, int M, int K, long long tot) {
  long long g = gtid(); if (g >= tot) return;
  int k = (int)(g % K); long long t = g / K;
  int mm = (int)(t % M); t /= M;
  int ch = (int)(t % (C + 3)); int b = (int)(t / (C + 3));
  int n = nidx[((size_t)b * M + mm) * K + k];
  float v;
  if (ch < 3)
    v = coords[((size_t)b * 3 + ch) * Np + n] -
        centers[((size_t)b * 3 + ch) * M + mm];
  else
    v = feat[((size_t)b * C + (ch - 3)) * Np + n];
  out[(((size_t)b * (C + 3) + ch) * M + mm) * K + k] = v;
}

__global__ void k_maxk(const float* __restrict__ x, float* __restrict__ y,
                       int O, int M, int K, long long tot) {
  long long g = gtid(); if (g >= tot) return;
  int mm = (int)(g % M); long long t = g / M;
  int o = (int)(t % O); int b = (int)(t / O);
  const float* p = x + (((size_t)b * O + o) * M + mm) * K;
  float mx = p[0];
  for (int k = 1; k < K; ++k) mx = fmaxf(mx, p[k]);
  y[((size_t)b * O + o) * M + mm] = mx;
}

// per-(b,dim) min/max reduction into mm[(b*3+d)*2 + {0,1}]
__global__ void k_minmax(const float* __restrict__ c, float* __restrict__ mm,
                         int Np) {
  __shared__ float smin[256], smax[256];
  int bd = blockIdx.x, t = threadIdx.x;
  const float* p = c + (size_t)bd * Np;
  float lo = 3e38f, hiv = -3e38f;
  for (int i = t; i < Np; i += 256) {
    float v = p[i]; lo = fminf(lo, v); hiv = fmaxf(hiv, v);
  }
  smin[t] = lo; smax[t] = hiv;
  __syncthreads();
  for (int s = 128; s > 0; s >>= 1) {
    if (t < s) {
      smin[t] = fminf(smin[t], smin[t + s]);
      smax[t] = fmaxf(smax[t], smax[t + s]);
    }
    __syncthreads();
  }
  if (t == 0) { mm[bd * 2] = smin[0]; mm[bd * 2 + 1] = smax[0]; }
}

// voxelize + per-axis flat index: flat3[a][b][n] = vox[u]*R + vox[v]
__global__ void k_voxflat(const float* __restrict__ coords,
                          const float* __restrict__ mm,
                          int* __restrict__ flat3, int R, int Np,
                          long long tot) {
  long long g = gtid(); if (g >= tot) return;
  int n = (int)(g % Np); int b = (int)(g / Np);
  int vox[3];
  for (int d = 0; d < 3; ++d) {
    float lo = mm[(b * 3 + d) * 2], hiv = mm[(b * 3 + d) * 2 + 1];
    float v = (coords[((size_t)b * 3 + d) * Np + n] - lo) /
              (hiv - lo + 1e-6f) * (float)(R - 1);
    int vi = (int)rintf(v);
    vi = vi < 0 ? 0 : (vi > R - 1 ? R - 1 : vi);
    vox[d] = vi;
  }
  flat3[((size_t)0 * Bn + b) * Np + n] = vox[1] * R + vox[2];
  flat3[((size_t)1 * Bn + b) * Np + n] = vox[0] * R + vox[2];
  flat3[((size_t)2 * Bn + b) * Np + n] = vox[0] * R + vox[1];
}

__global__ void k_scatter_cnt(const int* __restrict__ flat,
                              float* __restrict__ cnt, int Np, int R2,
                              long long tot) {
  long long g = gtid(); if (g >= tot) return;
  int n = (int)(g % Np); int b = (int)(g / Np);
  atomicAdd(&cnt[(size_t)b * R2 + flat[(size_t)b * Np + n]], 1.f);
}

__global__ void k_scatter_add(const float* __restrict__ feat,
                              const int* __restrict__ flat,
                              float* __restrict__ grid, int C, int Np, int R2,
                              long long tot) {
  long long g = gtid(); if (g >= tot) return;
  int c = (int)(g % C); long long t = g / C;
  int n = (int)(t % Np); int b = (int)(t / Np);
  int f = flat[(size_t)b * Np + n];
  atomicAdd(&grid[((size_t)b * R2 + f) * C + c],
            feat[((size_t)b * C + c) * Np + n]);
}

__global__ void k_gridnorm(float* __restrict__ grid,
                           const float* __restrict__ cnt, int C,
                           long long tot) {
  long long g = gtid(); if (g >= tot) return;
  grid[g] /= fmaxf(cnt[g / C], 1.f);
}

// grid [B,R,R,Ci] -> out [B,R,R,Co], 'SAME' 3x3, fused bn+relu
__global__ void k_conv3x3(const float* __restrict__ grid,
                          const float* __restrict__ w,
                          const float* __restrict__ gm,
                          const float* __restrict__ bt,
                          float* __restrict__ out,
                          int R, int Ci, int Co, long long tot) {
  long long g = gtid(); if (g >= tot) return;
  int co = (int)(g % Co); long long t = g / Co;
  int xq = (int)(t % R); t /= R;
  int yq = (int)(t % R); int b = (int)(t / R);
  float acc = 0.f;
  for (int ky = 0; ky < 3; ++ky) {
    int iy = yq + ky - 1; if (iy < 0 || iy >= R) continue;
    for (int kx = 0; kx < 3; ++kx) {
      int ix = xq + kx - 1; if (ix < 0 || ix >= R) continue;
      const float* gp = grid + (((size_t)b * R + iy) * R + ix) * Ci;
      const float* wp = w + (size_t)(ky * 3 + kx) * Ci * Co + co;
      for (int ci = 0; ci < Ci; ++ci) acc += gp[ci] * wp[(size_t)ci * Co];
    }
  }
  float r = acc * gm[co] + bt[co];
  out[(((size_t)b * R + yq) * R + xq) * Co + co] = fmaxf(r, 0.f);
}

// out[b,c,n] += grid2[b, flat[b,n], c]
__global__ void k_gatherback(const float* __restrict__ grid2,
                             const int* __restrict__ flat,
                             float* __restrict__ out,
                             int Co, int Np, int R2, long long tot) {
  long long g = gtid(); if (g >= tot) return;
  int n = (int)(g % Np); long long t = g / Np;
  int c = (int)(t % Co); int b = (int)(t / Co);
  out[((size_t)b * Co + c) * Np + n] +=
      grid2[((size_t)b * R2 + flat[(size_t)b * Np + n]) * Co + c];
}

__global__ void k_chanmean(const float* __restrict__ x, float* __restrict__ s,
                           int C, int Np, long long tot) {
  long long g = gtid(); if (g >= tot) return;
  int c = (int)(g % C); int b = (int)(g / C);
  const float* p = x + ((size_t)b * C + c) * Np;
  float a = 0.f;
  for (int n = 0; n < Np; ++n) a += p[n];
  s[(size_t)b * C + c] = a / (float)Np;
}

// Squeeze-excite FCs; one block per batch
__global__ void k_se(const float* __restrict__ s, const float* __restrict__ w1,
                     const float* __restrict__ b1, const float* __restrict__ w2,
                     const float* __restrict__ b2, float* __restrict__ sig,
                     int C, int H) {
  __shared__ float hb[64];
  int b = blockIdx.x, t = threadIdx.x;
  if (t < H) {
    float a = b1[t];
    for (int c = 0; c < C; ++c) a += w1[(size_t)t * C + c] * s[(size_t)b * C + c];
    hb[t] = fmaxf(a, 0.f);
  }
  __syncthreads();
  if (t < C) {
    float a = b2[t];
    for (int j = 0; j < H; ++j) a += w2[(size_t)t * H + j] * hb[j];
    sig[(size_t)b * C + t] = 1.f / (1.f + expf(-a));
  }
}

__global__ void k_scale(float* __restrict__ x, const float* __restrict__ sig,
                        int C, int Np, long long tot) {
  long long g = gtid(); if (g >= tot) return;
  long long t = g / Np;
  int c = (int)(t % C); int b = (int)(t / C);
  x[g] *= sig[(size_t)b * C + c];
}

// 3 nearest coarse points + inverse-distance weights
__global__ void k_3nn(const float* __restrict__ cd, const float* __restrict__ cc,
                      int* __restrict__ idx3, float* __restrict__ w3,
                      int Nd, int Mc, long long tot) {
  long long g = gtid(); if (g >= tot) return;
  int n = (int)(g % Nd); int b = (int)(g / Nd);
  float px = cd[((size_t)b * 3 + 0) * Nd + n];
  float py = cd[((size_t)b * 3 + 1) * Nd + n];
  float pz = cd[((size_t)b * 3 + 2) * Nd + n];
  const float* qx = cc + (size_t)b * 3 * Mc;
  const float* qy = qx + Mc;
  const float* qz = qy + Mc;
  float d0 = 3e38f, d1 = 3e38f, d2v = 3e38f;
  int i0 = 0, i1 = 0, i2 = 0;
  for (int j = 0; j < Mc; ++j) {
    float dx = qx[j] - px, dy = qy[j] - py, dz = qz[j] - pz;
    float d = dx * dx + dy * dy + dz * dz;
    if (d < d0) { d2v = d1; i2 = i1; d1 = d0; i1 = i0; d0 = d; i0 = j; }
    else if (d < d1) { d2v = d1; i2 = i1; d1 = d; i1 = j; }
    else if (d < d2v) { d2v = d; i2 = j; }
  }
  float w0 = 1.f / (fmaxf(d0, 0.f) + 1e-8f);
  float w1 = 1.f / (fmaxf(d1, 0.f) + 1e-8f);
  float w2 = 1.f / (fmaxf(d2v, 0.f) + 1e-8f);
  float s = w0 + w1 + w2;
  size_t o = ((size_t)b * Nd + n) * 3;
  idx3[o] = i0; idx3[o + 1] = i1; idx3[o + 2] = i2;
  w3[o] = w0 / s; w3[o + 1] = w1 / s; w3[o + 2] = w2 / s;
}

__global__ void k_interp(const float* __restrict__ feat,
                         const int* __restrict__ idx3,
                         const float* __restrict__ w3,
                         float* __restrict__ out,
                         int C, int Nd, int Mc, long long tot) {
  long long g = gtid(); if (g >= tot) return;
  int n = (int)(g % Nd); long long t = g / Nd;
  int c = (int)(t % C); int b = (int)(t / C);
  size_t o = ((size_t)b * Nd + n) * 3;
  const float* f = feat + ((size_t)b * C + c) * Mc;
  out[((size_t)b * C + c) * Nd + n] =
      f[idx3[o]] * w3[o] + f[idx3[o + 1]] * w3[o + 1] + f[idx3[o + 2]] * w3[o + 2];
}

// ---------------------------------------------------------------------------
// Host orchestration
// ---------------------------------------------------------------------------
struct PP {
  const float *axb[3], *axg[3], *axw[3];
  const float *ptb, *ptg, *ptw;
  const float *seb1, *seb2, *sew1, *sew2;
};
struct MLPL { const float *b, *g, *w; };

static inline int nblk(long long total) {
  return (int)((total + WG - 1) / WG);
}

extern "C" void kernel_launch(void* const* d_in, const int* in_sizes, int n_in,
                              void* d_out, int out_size, void* d_ws,
                              size_t ws_size, hipStream_t stream) {
  (void)in_sizes; (void)out_size; (void)ws_size;

  // ---- parameter walker (jax tree-flatten order: dict keys sorted) --------
  int cursor = 1;
  auto nxt = [&]() -> const float* {
    int i = cursor < n_in ? cursor : (n_in - 1);
    ++cursor;
    return (const float*)d_in[i];
  };
  auto readPP = [&](PP& p) {
    for (int a = 0; a < 3; ++a) { p.axb[a] = nxt(); p.axg[a] = nxt(); p.axw[a] = nxt(); }
    p.ptb = nxt(); p.ptg = nxt(); p.ptw = nxt();
    p.seb1 = nxt(); p.seb2 = nxt(); p.sew1 = nxt(); p.sew2 = nxt();
  };
  auto readMLP = [&](MLPL& l) { l.b = nxt(); l.g = nxt(); l.w = nxt(); };

  // 'cls' first
  MLPL cls_l[3]; const float *cls_bias, *cls_wf;
  for (int i = 0; i < 3; ++i) readMLP(cls_l[i]);
  cls_bias = nxt(); cls_wf = nxt();
  // 'fp'
  const int fp_nconv[4] = {1, 1, 2, 1};
  const int fp_nmlp[4]  = {2, 2, 2, 3};
  PP fp_pp[4][2]; MLPL fp_mlp[4][3];
  for (int j = 0; j < 4; ++j) {
    for (int c = 0; c < fp_nconv[j]; ++c) readPP(fp_pp[j][c]);
    for (int l = 0; l < fp_nmlp[j]; ++l) readMLP(fp_mlp[j][l]);
  }
  // 'sa'
  const int sa_nconv[4] = {2, 3, 3, 0};
  const int sa_nmlp[4]  = {2, 2, 2, 3};
  PP sa_pp[4][3]; MLPL sa_mlp[4][3];
  for (int i = 0; i < 4; ++i) {
    for (int c = 0; c < sa_nconv[i]; ++c) readPP(sa_pp[i][c]);
    for (int l = 0; l < sa_nmlp[i]; ++l) readMLP(sa_mlp[i][l]);
  }

  // ---- workspace arena ----------------------------------------------------
  uint8_t* wp = (uint8_t*)d_ws;
  size_t off = 0;
  auto alF = [&](size_t n) -> float* {
    float* p = (float*)(wp + off);
    off += ((n * 4 + 255) & ~(size_t)255);
    return p;
  };
  auto alI = [&](size_t n) -> int* {
    int* p = (int*)(wp + off);
    off += ((n * 4 + 255) & ~(size_t)255);
    return p;
  };
  auto alU = [&](size_t n) -> unsigned short* {
    unsigned short* p = (unsigned short*)(wp + off);
    off += ((n * 2 + 255) & ~(size_t)255);
    return p;
  };

  const size_t BIGN = (size_t)Bn * 68 * 32768;  // ping-pong activation pool
  float* big0 = alF(BIGN);
  float* big1 = alF(BIGN);
  float* big2 = alF(BIGN);
  const int L_pts[5] = {8192, 1024, 256, 64, 16};
  float* coordsL[5];
  for (int i = 0; i < 5; ++i) coordsL[i] = alF((size_t)Bn * 3 * L_pts[i]);
  float* feat0  = alF((size_t)Bn * 6 * NPTS);
  float* initf  = alF((size_t)Bn * 3 * NPTS);
  float* onehot = alF((size_t)Bn * 16 * NPTS);
  float* saOut[4] = { alF((size_t)Bn * 64 * 1024), alF((size_t)Bn * 128 * 256),
                      alF((size_t)Bn * 256 * 64),  alF((size_t)Bn * 512 * 16) };
  float* gridbuf  = alF((size_t)Bn * 65536);
  float* gridbuf2 = alF((size_t)Bn * 65536);
  float* cntbuf   = alF((size_t)Bn * 1024);
  float* mmbuf    = alF((size_t)Bn * 6);
  float* sbuf     = alF((size_t)Bn * 256);
  float* sigbuf   = alF((size_t)Bn * 256);
  float* w3buf    = alF((size_t)Bn * NPTS * 3);
  unsigned short* WbfBuf = alU((size_t)512 * 800);          // padded bf16 W
  unsigned short* XtBuf  = alU((size_t)Bn * 32768 * 68);    // padded bf16 X^T
  int* flat3  = alI((size_t)3 * Bn * NPTS);
  int* fpsidx = alI((size_t)Bn * 1024);
  int* nidx   = alI((size_t)Bn * 1024 * 32);
  int* idx3   = alI((size_t)Bn * NPTS * 3);

  // ---- launch helpers -----------------------------------------------------
  auto Z = [&](float* p, long long n) {
    k_zero<<<nblk(n), WG, 0, stream>>>(p, n);
  };
  auto GEMM = [&](const float* W, const float* X, const float* G,
                  const float* Bv, float* Y, int O, int C, int Nn, int mode) {
    int Opad = (O + 63) & ~63;
    int Cpad = (C + 31) & ~31;
    int Npad = (Nn + 63) & ~63;
    { long long t = (long long)Opad * Cpad;
      k_castW<<<nblk(t), WG, 0, stream>>>(W, WbfBuf, O, C, Cpad, t); }
    { long long t = (long long)Bn * Npad * Cpad;
      k_castXT<<<nblk(t), WG, 0, stream>>>(X, XtBuf, C, Nn, Cpad, Npad, t); }
    dim3 gr(Npad / 64, Opad / 64, Bn);
    k_gemm<<<gr, 256, 0, stream>>>(WbfBuf, XtBuf, G, Bv, Y, O, C, Nn,
                                   Cpad, Npad, mode);
  };
  auto PLACE = [&](const float* src, float* dst, int Cs, int cs0, int Cd,
                   int cd0, int Cc, int Nn) {
    long long tot = (long long)Bn * Cc * Nn;
    k_place<<<nblk(tot), WG, 0, stream>>>(src, dst, Cs, cs0, Cd, cd0, Cc, Nn, tot);
  };
  auto VOX = [&](const float* cds, int Np, int R) {
    k_minmax<<<Bn * 3, 256, 0, stream>>>(cds, mmbuf, Np);
    long long tot = (long long)Bn * Np;
    k_voxflat<<<nblk(tot), WG, 0, stream>>>(cds, mmbuf, flat3, R, Np, tot);
  };
  auto PPC = [&](const PP& p, const float* fin, float* fo, int ci, int co,
                 int Np, int R) {
    GEMM(p.ptw, fin, p.ptg, p.ptb, fo, co, ci, Np, 1);
    int R2 = R * R;
    for (int a = 0; a < 3; ++a) {
      const int* fl = flat3 + (size_t)a * Bn * Np;
      Z(gridbuf, (long long)Bn * R2 * ci);
      Z(cntbuf, (long long)Bn * R2);
      long long t1 = (long long)Bn * Np;
      k_scatter_cnt<<<nblk(t1), WG, 0, stream>>>(fl, cntbuf, Np, R2, t1);
      long long t2 = (long long)Bn * Np * ci;
      k_scatter_add<<<nblk(t2), WG, 0, stream>>>(fin, fl, gridbuf, ci, Np, R2, t2);
      long long t3 = (long long)Bn * R2 * ci;
      k_gridnorm<<<nblk(t3), WG, 0, stream>>>(gridbuf, cntbuf, ci, t3);
      long long t4 = (long long)Bn * R2 * co;
      k_conv3x3<<<nblk(t4), WG, 0, stream>>>(gridbuf, p.axw[a], p.axg[a],
                                             p.axb[a], gridbuf2, R, ci, co, t4);
      long long t5 = (long long)Bn * co * Np;
      k_gatherback<<<nblk(t5), WG, 0, stream>>>(gridbuf2, fl, fo, co, Np, R2, t5);
    }
    int H = (co / 8 < 4) ? 4 : co / 8;
    long long tc = (long long)Bn * co;
    k_chanmean<<<nblk(tc), WG, 0, stream>>>(fo, sbuf, co, Np, tc);
    k_se<<<Bn, 256, 0, stream>>>(sbuf, p.sew1, p.seb1, p.sew2, p.seb2, sigbuf,
                                 co, H);
    long long ts = (long long)Bn * co * Np;
    k_scale<<<nblk(ts), WG, 0, stream>>>(fo, sigbuf, co, Np, ts);
  };
  auto pick = [&](const void* a, const void* b) -> float* {
    float* pool[3] = { big0, big1, big2 };
    for (int i = 0; i < 3; ++i)
      if ((void*)pool[i] != a && (void*)pool[i] != b) return pool[i];
    return big0;
  };

  // ---- split inputs: [B,22,8192] -> coords / feat(6) / init(3) / onehot(16)
  const float* inp = (const float*)d_in[0];
  PLACE(inp, coordsL[0], 22, 0, 3, 0, 3, NPTS);
  PLACE(inp, feat0,      22, 0, 6, 0, 6, NPTS);
  PLACE(inp, initf,      22, 3, 3, 0, 3, NPTS);
  PLACE(inp, onehot,     22, 6, 16, 0, 16, NPTS);

  // ---- SA stages ----------------------------------------------------------
  const int   sa_convco[4] = {32, 64, 128, 0};
  const int   sa_R[4]      = {32, 16, 8, 0};
  const int   sa_M[4]      = {1024, 256, 64, 16};
  const float sa_r2[4]     = {0.01f, 0.04f, 0.16f, 0.64f};
  const int   sa_dims[4][4] = {{35, 32, 64, 0}, {67, 64, 128, 0},
                               {131, 128, 256, 0}, {259, 256, 256, 512}};

  const float* fcur = feat0;
  int ccur = 6;
  for (int i = 0; i < 4; ++i) {
    int Np = L_pts[i];
    if (sa_nconv[i] > 0) {
      VOX(coordsL[i], Np, sa_R[i]);
      int ci = ccur;
      for (int c = 0; c < sa_nconv[i]; ++c) {
        float* fo = pick(fcur, nullptr);
        PPC(sa_pp[i][c], fcur, fo, ci, sa_convco[i], Np, sa_R[i]);
        fcur = fo; ci = sa_convco[i];
      }
      ccur = ci;
    }
    int M = sa_M[i];
    k_fps<<<Bn, 256, 0, stream>>>(coordsL[i], fpsidx, Np, M);
    { long long t = (long long)Bn * 3 * M;
      k_gather_cols<<<nblk(t), WG, 0, stream>>>(coordsL[i], fpsidx,
                                                coordsL[i + 1], 3, Np, M, t); }
    { long long t = (long long)Bn * M;
      k_ballquery<<<nblk(t), WG, 0, stream>>>(coordsL[i], coordsL[i + 1], nidx,
                                              sa_r2[i], Np, M, 32, t); }
    float* grp = pick(fcur, nullptr);
    { long long t = (long long)Bn * (ccur + 3) * M * 32;
      k_group<<<nblk(t), WG, 0, stream>>>(fcur, coordsL[i], coordsL[i + 1],
                                          nidx, grp, ccur, Np, M, 32, t); }
    const float* x = grp; int cin = ccur + 3; int NMK = M * 32;
    for (int l = 0; l < sa_nmlp[i]; ++l) {
      float* y = pick(x, nullptr);
      GEMM(sa_mlp[i][l].w, x, sa_mlp[i][l].g, sa_mlp[i][l].b, y,
           sa_dims[i][l + 1], cin, NMK, 1);
      x = y; cin = sa_dims[i][l + 1];
    }
    { long long t = (long long)Bn * cin * M;
      k_maxk<<<nblk(t), WG, 0, stream>>>(x, saOut[i], cin, M, 32, t); }
    fcur = saOut[i]; ccur = cin;
  }

  // ---- FP stages ----------------------------------------------------------
  const float* skipP[4] = { saOut[2], saOut[1], saOut[0], initf };
  const int skipC[4]      = {256, 128, 64, 3};
  const int fp_dims[4][4] = {{768, 256, 256, 0}, {384, 256, 256, 0},
                             {320, 256, 128, 0}, {131, 128, 128, 64}};
  const int fp_convco[4]  = {256, 256, 128, 64};
  const int fp_R[4]       = {8, 8, 16, 32};

  for (int j = 0; j < 4; ++j) {
    int Nd = L_pts[3 - j];
    int Mc = L_pts[4 - j];
    const float* cdense = coordsL[3 - j];
    const float* ccoarse = coordsL[4 - j];
    { long long t = (long long)Bn * Nd;
      k_3nn<<<nblk(t), WG, 0, stream>>>(cdense, ccoarse, idx3, w3buf, Nd, Mc, t); }
    float* t_interp = pick(fcur, nullptr);
    { long long t = (long long)Bn * ccur * Nd;
      k_interp<<<nblk(t), WG, 0, stream>>>(fcur, idx3, w3buf, t_interp, ccur,
                                           Nd, Mc, t); }
    int cc = ccur + skipC[j];
    float* t_cat = pick(t_interp, nullptr);
    PLACE(t_interp, t_cat, ccur, 0, cc, 0, ccur, Nd);
    PLACE(skipP[j], t_cat, skipC[j], 0, cc, ccur, skipC[j], Nd);
    const float* x = t_cat; int cin = cc;
    for (int l = 0; l < fp_nmlp[j]; ++l) {
      float* y = pick(x, nullptr);
      GEMM(fp_mlp[j][l].w, x, fp_mlp[j][l].g, fp_mlp[j][l].b, y,
           fp_dims[j][l + 1], cin, Nd, 1);
      x = y; cin = fp_dims[j][l + 1];
    }
    VOX(cdense, Nd, fp_R[j]);
    for (int c = 0; c < fp_nconv[j]; ++c) {
      float* fo = pick(x, nullptr);
      PPC(fp_pp[j][c], x, fo, cin, fp_convco[j], Nd, fp_R[j]);
      x = fo; cin = fp_convco[j];
    }
    fcur = x; ccur = cin;
  }

  // ---- classifier head ----------------------------------------------------
  int cc = ccur + 16;  // 64 + 16 = 80
  float* t_cat = pick(fcur, nullptr);
  PLACE(fcur, t_cat, ccur, 0, cc, 0, ccur, NPTS);
  PLACE(onehot, t_cat, 16, 0, cc, ccur, 16, NPTS);
  const int cls_dims[3] = {256, 256, 128};
  const float* x = t_cat; int cin = cc;
  for (int l = 0; l < 3; ++l) {
    float* y = pick(x, nullptr);
    GEMM(cls_l[l].w, x, cls_l[l].g, cls_l[l].b, y, cls_dims[l], cin, NPTS, 1);
    x = y; cin = cls_dims[l];
  }
  GEMM(cls_wf, x, nullptr, cls_bias, (float*)d_out, 50, cin, NPTS, 2);
}
